// PointNetSAModule_4148938408194
// MI455X (gfx1250) — matrix operand order repaired and checked by reading, not compiled, not run on hardware
//
#include <hip/hip_runtime.h>
#include <cstdint>
#include <cstddef>

typedef __attribute__((ext_vector_type(16))) __bf16 v16bf;
typedef __attribute__((ext_vector_type(8)))  float  v8f;
typedef unsigned int __attribute__((ext_vector_type(4))) u32x4;
typedef int __attribute__((ext_vector_type(4))) i32x4;
typedef int __attribute__((ext_vector_type(8))) i32x8;

#define B_     16
#define N_     4096
#define M_     1024
#define K_     32
#define MK_    (M_ * K_)        // 32768
#define NTOT_  (B_ * MK_)       // 524288
#define EPS_   1e-5f
#define R2_    0.04f            // RADIUS^2

// ---- Tensor Data Mover availability probes (compile-safe) ------------------
#if __has_builtin(__builtin_amdgcn_tensor_load_to_lds)
#define HAVE_TDM 1
#if defined(__clang_major__) && __clang_major__ >= 23
#define TDM_LOAD(g0, g1) \
    __builtin_amdgcn_tensor_load_to_lds((g0), (g1), (i32x4)0, (i32x4)0, (i32x8)0, 0)
#else
#define TDM_LOAD(g0, g1) \
    __builtin_amdgcn_tensor_load_to_lds((g0), (g1), (i32x4)0, (i32x4)0, 0)
#endif
#endif

#if __has_builtin(__builtin_amdgcn_s_wait_tensorcnt)
#define WAIT_TENSOR() __builtin_amdgcn_s_wait_tensorcnt(0)
#else
#define WAIT_TENSOR() asm volatile("s_wait_tensorcnt 0x0" ::: "memory")
#endif

// ---------------------------------------------------------------------------
// 1) Furthest point sampling: one 1024-thread block per batch.
// ---------------------------------------------------------------------------
__global__ __launch_bounds__(1024) void fps_kernel(const float* __restrict__ coords,
                                                   float* __restrict__ centers) {
    __shared__ float px[N_], py[N_], pz[N_];
    __shared__ float rv[32];
    __shared__ int   ri[32];
    __shared__ int   s_last;
    const int b = blockIdx.x, tid = threadIdx.x;
    const int lane = tid & 31, wave = tid >> 5;
    const float* cb = coords + (size_t)b * 3 * N_;
    for (int n = tid; n < N_; n += 1024) {
        px[n] = cb[n]; py[n] = cb[N_ + n]; pz[n] = cb[2 * N_ + n];
    }
    float dl[4];
#pragma unroll
    for (int i = 0; i < 4; ++i) dl[i] = 1e10f;
    __syncthreads();

    float* cbo = centers + (size_t)b * 3 * M_;
    int last = 0;
    if (tid == 0) { cbo[0] = px[0]; cbo[M_] = py[0]; cbo[2 * M_] = pz[0]; }

    for (int j = 1; j < M_; ++j) {
        const float lx = px[last], ly = py[last], lz = pz[last];
        float best = -1.f; int bi = 0;
#pragma unroll
        for (int i = 0; i < 4; ++i) {
            const int n = tid + i * 1024;
            const float dx = px[n] - lx, dy = py[n] - ly, dz = pz[n] - lz;
            const float d = dx * dx + dy * dy + dz * dz;
            const float nd = fminf(dl[i], d);
            dl[i] = nd;
            if (nd > best) { best = nd; bi = n; }
        }
#pragma unroll
        for (int off = 16; off; off >>= 1) {
            const float ov = __shfl_xor(best, off);
            const int   oi = __shfl_xor(bi, off);
            if (ov > best || (ov == best && oi < bi)) { best = ov; bi = oi; }
        }
        if (lane == 0) { rv[wave] = best; ri[wave] = bi; }
        __syncthreads();
        if (wave == 0) {
            best = rv[lane]; bi = ri[lane];
#pragma unroll
            for (int off = 16; off; off >>= 1) {
                const float ov = __shfl_xor(best, off);
                const int   oi = __shfl_xor(bi, off);
                if (ov > best || (ov == best && oi < bi)) { best = ov; bi = oi; }
            }
            if (lane == 0) {
                s_last = bi;
                cbo[j] = px[bi]; cbo[M_ + j] = py[bi]; cbo[2 * M_ + j] = pz[bi];
            }
        }
        __syncthreads();
        last = s_last;
    }
}

// ---------------------------------------------------------------------------
// 2) Ball query: one wave32 per center; ballot + prefix popcount.
// ---------------------------------------------------------------------------
__global__ __launch_bounds__(256) void ballq_kernel(const float* __restrict__ coords,
                                                    const float* __restrict__ centers,
                                                    int* __restrict__ nidx) {
    __shared__ int sidx[8][K_];
    const int wave = threadIdx.x >> 5, lane = threadIdx.x & 31;
    const int cid = blockIdx.x * 8 + wave;          // < B*M
    const int b = cid / M_, m = cid % M_;
    const float* cb = coords + (size_t)b * 3 * N_;
    const float cx = centers[(size_t)b * 3 * M_ + m];
    const float cy = centers[(size_t)b * 3 * M_ + M_ + m];
    const float cz = centers[(size_t)b * 3 * M_ + 2 * M_ + m];
    int cnt = 0;
    for (int base = 0; base < N_ && cnt < K_; base += 32) {
        const int n = base + lane;
        const float dx = cb[n] - cx, dy = cb[N_ + n] - cy, dz = cb[2 * N_ + n] - cz;
        const bool in = (dx * dx + dy * dy + dz * dz) < R2_;
        const unsigned bal = (unsigned)__ballot(in);
        const int pos = cnt + __popc(bal & ((1u << lane) - 1u));
        if (in && pos < K_) sidx[wave][pos] = n;
        cnt += __popc(bal);
    }
    __syncthreads();
    const int valid = cnt < K_ ? cnt : K_;
    const int first = valid > 0 ? sidx[wave][0] : 0;
    nidx[(size_t)cid * K_ + lane] = (lane < valid) ? sidx[wave][lane] : first;
}

// ---------------------------------------------------------------------------
// 3a) Transpose features to point-major bf16: feat[B,64,N] -> featT[B,N,64].
// ---------------------------------------------------------------------------
__global__ __launch_bounds__(256) void featT_kernel(const float* __restrict__ feat,
                                                    __bf16* __restrict__ featT) {
    const int gid = blockIdx.x * 256 + threadIdx.x;     // < B*N*8
    const int n   = gid & (B_ * N_ - 1);
    const int seg = gid >> 16;                          // 0..7
    const int b = n >> 12, nn = n & (N_ - 1);
    union { uint4 u; __bf16 h[8]; } o;
    const float* fb = feat + ((size_t)b * 64 + seg * 8) * N_ + nn;
#pragma unroll
    for (int q = 0; q < 8; ++q) o.h[q] = (__bf16)fb[(size_t)q * N_];
    *(uint4*)(featT + (size_t)n * 64 + seg * 8) = o.u;
}

// ---------------------------------------------------------------------------
// 3b) Gather -> x0 bf16 [NTOT][96] point-major: ch 0..63 = features,
//     64..66 = relative coords, 67..95 = zero K-padding. All uint4 I/O.
// ---------------------------------------------------------------------------
__global__ __launch_bounds__(256) void gather_kernel(const __bf16* __restrict__ featT,
                                                     const float* __restrict__ coords,
                                                     const float* __restrict__ centers,
                                                     const int* __restrict__ nidx,
                                                     __bf16* __restrict__ x0) {
    const int n = blockIdx.x * 256 + threadIdx.x;   // < NTOT
    const int b = n >> 15;
    const int m = (n >> 5) & (M_ - 1);
    const int p = nidx[n];
    union { uint4 u[12]; __bf16 h[96]; } row;
    const uint4* fp = (const uint4*)(featT + ((size_t)b * N_ + p) * 64);
#pragma unroll
    for (int q = 0; q < 8; ++q) row.u[q] = fp[q];
    const float* cb = coords + (size_t)b * 3 * N_;
    row.h[64] = (__bf16)(cb[p]          - centers[(size_t)b * 3 * M_ + m]);
    row.h[65] = (__bf16)(cb[N_ + p]     - centers[(size_t)b * 3 * M_ + M_ + m]);
    row.h[66] = (__bf16)(cb[2 * N_ + p] - centers[(size_t)b * 3 * M_ + 2 * M_ + m]);
#pragma unroll
    for (int t = 67; t < 96; ++t) row.h[t] = (__bf16)0.f;
    uint4* xp = (uint4*)(x0 + (size_t)n * 96);
#pragma unroll
    for (int q = 0; q < 12; ++q) xp[q] = row.u[q];
}

// ---------------------------------------------------------------------------
// 4) GEMM (1x1 conv) with bf16 WMMA + deterministic GroupNorm partial sums.
//    Activations point-major [n][CIP]; per block one 256-point slab staged to
//    LDS by the Tensor Data Mover (fallback: vector copy loop). nt loop fully
//    unrolled with wave-uniform base so all LDS traffic folds to immediate
//    offsets and WMMAs pipeline against the next tile's loads.
// ---------------------------------------------------------------------------
template <int COUT, int CIP, int CIN>
__global__ __launch_bounds__(256) void gemm_gn_kernel(const __bf16* __restrict__ x,
                                                      const float* __restrict__ w,
                                                      const float* __restrict__ bias,
                                                      __bf16* __restrict__ y,
                                                      float* __restrict__ partials) {
    constexpr int TCO = COUT / 16;       // co tiles: 4 or 8
    constexpr int WPC = 8 / TCO;         // waves per co-tile: 2 or 1
    constexpr int NTW = 16 / WPC;        // n-tiles per wave: 8 or 16
    constexpr int KS  = CIP / 32;        // k-steps: 3 or 2
    constexpr int XSB = CIP * 256 * 2;   // x slab bytes (at LDS offset 0)
    constexpr int WB  = COUT * CIP * 2;

    extern __shared__ __attribute__((aligned(16))) char smem[];
    __bf16* xs    = (__bf16*)smem;                       // [256][CIP]
    __bf16* wlds  = (__bf16*)(smem + XSB);               // [COUT][CIP]
    float*  blds  = (float*)(smem + XSB + WB);
    float*  hstat = (float*)(smem + XSB + WB + COUT * 4);

    const int tid = threadIdx.x, lane = tid & 31;
    const int wave = __builtin_amdgcn_readfirstlane(tid >> 5);   // wave-uniform scalar
    const int cot = wave % TCO, sub = wave / TCO;
    const int n0 = blockIdx.x * 256;

    // ---- stage x slab (contiguous 256*CIP bf16) into LDS ----
#if defined(HAVE_TDM)
    if (wave == 0) {
        const unsigned long long ga =
            (unsigned long long)(const void*)x + (unsigned long long)n0 * (CIP * 2);
        const unsigned elems = (unsigned)(CIP * 256);    // 2-byte units
        u32x4 g0;
        g0[0] = 1u;                                      // count=1, user mode
        g0[1] = 0u;                                      // lds_addr = 0 (xs at base)
        g0[2] = (unsigned)(ga & 0xffffffffu);            // global_addr[31:0]
        g0[3] = (unsigned)((ga >> 32) & 0x01ffffffu) | 0x80000000u;  // addr[56:32] | type=2
        i32x8 g1;
        g1[0] = (1 << 16);                               // data_size=1 (2B), no flags
        g1[1] = (int)((elems & 0xffffu) << 16);          // tensor_dim0[15:0]
        g1[2] = (int)((elems >> 16) | (1u << 16));       // tensor_dim0[31:16] | tensor_dim1=1
        g1[3] = (int)((elems & 0xffffu) << 16);          // tile_dim0 = elems
        g1[4] = 1;                                       // tile_dim1=1, tile_dim2=0
        g1[5] = (int)elems;                              // tensor_dim0_stride lo
        g1[6] = 0;
        g1[7] = 0;
        TDM_LOAD(g0, g1);
        WAIT_TENSOR();
    }
#else
    for (int i = tid; i < CIP * 32; i += 256)
        ((uint4*)xs)[i] = *((const uint4*)(x + (size_t)n0 * CIP) + i);
#endif

    // ---- stage weights (with layer-1 channel permutation) + bias ----
    for (int i = tid; i < COUT * CIP; i += 256) {
        const int co = i / CIP, ci = i % CIP;
        float v;
        if (CIN == 67)          // x0 channel order: [feat 0..63, coord 0..2, pad]
            v = (ci < 64) ? w[co * 67 + 3 + ci] : ((ci < 67) ? w[co * 67 + (ci - 64)] : 0.f);
        else
            v = w[co * CIN + ci];
        wlds[i] = (__bf16)v;
    }
    for (int i = tid; i < COUT; i += 256) blds[i] = bias[i];
    __syncthreads();

    typedef union { uint4 u[2]; v16bf v; } frag_u;

    // A fragments resident in registers: lane<16 -> K {0..7,16..23} of k-step
    v16bf afr[KS];
    const int mrow = cot * 16 + (lane & 15);
    const int koffA = (lane < 16) ? 0 : 8;
#pragma unroll
    for (int ks = 0; ks < KS; ++ks) {
        frag_u f;
        f.u[0] = *(const uint4*)(wlds + mrow * CIP + ks * 32 + koffA);
        f.u[1] = *(const uint4*)(wlds + mrow * CIP + ks * 32 + koffA + 16);
        afr[ks] = f.v;
    }

    const int col = lane & 15;
    const int koffB = (lane < 16) ? 0 : 8;
    const int rbase = cot * 16 + ((lane < 16) ? 0 : 8);

    // bias for this lane's 8 consecutive channels (loop invariant)
    float bb[8];
#pragma unroll
    for (int r = 0; r < 8; ++r) bb[r] = blds[rbase + r];

    const __bf16* xbase = xs + (sub * NTW * 16 + col) * CIP + koffB;
    __bf16* ybase = y + ((size_t)n0 + sub * NTW * 16 + col) * COUT + rbase;
    float s1 = 0.f, s2 = 0.f;

#pragma unroll
    for (int nti = 0; nti < NTW; ++nti) {      // constexpr trip: fully unrolled,
        v8f c = {0.f, 0.f, 0.f, 0.f, 0.f, 0.f, 0.f, 0.f};  // LDS offsets immediate
        const __bf16* xrow = xbase + nti * 16 * CIP;
#pragma unroll
        for (int ks = 0; ks < KS; ++ks) {
            frag_u f;                           // K {0..7,16..23} (+8 hi half): 2x ds_load_b128
            f.u[0] = *(const uint4*)(xrow + ks * 32);
            f.u[1] = *(const uint4*)(xrow + ks * 32 + 16);
            c = __builtin_amdgcn_wmma_f32_16x16x32_bf16(
                    false, afr[ks], false, f.v, (short)0, c, false, false);
        }
        union { uint4 u; __bf16 h[8]; } o;
#pragma unroll
        for (int r = 0; r < 8; ++r) {
            const float v = c[r] + bb[r];
            s1 += v; s2 += v * v;
            o.h[r] = (__bf16)v;
        }
        *(uint4*)(ybase + (size_t)nti * 16 * COUT) = o.u;   // 8 consecutive channels
    }

    // deterministic stats: reduce within 16-lane halves, fixed-order combine
#pragma unroll
    for (int off = 8; off; off >>= 1) { s1 += __shfl_xor(s1, off); s2 += __shfl_xor(s2, off); }
    if ((lane & 15) == 0) {
        const int h = wave * 2 + (lane >> 4);
        hstat[h * 2] = s1; hstat[h * 2 + 1] = s2;
    }
    __syncthreads();
    if (tid < 16) {
        const int g = tid >> 1, f = tid & 1;
        float s = 0.f;
#pragma unroll
        for (int h = 0; h < 16; ++h) {
            const int wv = h >> 1;
            const int gh = (COUT == 64) ? ((wv % TCO) * 2 + (h & 1)) : wv;
            if (gh == g) s += hstat[h * 2 + f];
        }
        partials[(size_t)blockIdx.x * 16 + tid] = s;
    }
}

// Fixed-order finalize: 128 threads = B*GROUPS; 128 blocks per batch.
__global__ void gn_finalize_kernel(const float* __restrict__ partials,
                                   float* __restrict__ stats, float cnt) {
    const int t = threadIdx.x;        // b*8 + g
    const int b = t >> 3, g = t & 7;
    float s = 0.f, ss = 0.f;
    const int base = b * 128;
    for (int q = 0; q < 128; ++q) {
        s  += partials[(size_t)(base + q) * 16 + g * 2];
        ss += partials[(size_t)(base + q) * 16 + g * 2 + 1];
    }
    const float mean = s / cnt;
    const float var = ss / cnt - mean * mean;
    stats[b * 16 + g * 2]     = mean;
    stats[b * 16 + g * 2 + 1] = rsqrtf(var + EPS_);
}

// GN + SiLU in place (layers 1 & 2), point-major [n][cout], 8 bf16/thread.
__global__ __launch_bounds__(256) void gn_apply_kernel(__bf16* __restrict__ x,
                                                       const float* __restrict__ stats,
                                                       const float* __restrict__ gamma,
                                                       const float* __restrict__ beta,
                                                       int cout, int chPer) {
    const size_t i8 = (size_t)blockIdx.x * 256 + threadIdx.x;
    const size_t e0 = i8 * 8;
    const int c0 = (int)(e0 % cout);
    const int n  = (int)(e0 / cout);
    const int b = n >> 15;
    const int g = c0 / chPer;
    const float mean = stats[b * 16 + g * 2], inv = stats[b * 16 + g * 2 + 1];
    union { uint4 u; __bf16 h[8]; } uu;
    uu.u = ((const uint4*)x)[i8];
#pragma unroll
    for (int t = 0; t < 8; ++t) {
        float v = ((float)uu.h[t] - mean) * inv * gamma[c0 + t] + beta[c0 + t];
        v = v / (1.f + __expf(-v));
        uu.h[t] = (__bf16)v;
    }
    ((uint4*)x)[i8] = uu.u;
}

// Layer-3 GN + SiLU fused with max over K=32 neighbors; y3 is [n][128],
// threads c-fastest so the k-strided loads coalesce across the wave.
__global__ __launch_bounds__(256) void gn_silu_max_kernel(const __bf16* __restrict__ y,
                                                          const float* __restrict__ stats,
                                                          const float* __restrict__ gamma,
                                                          const float* __restrict__ beta,
                                                          float* __restrict__ out) {
    const int t = blockIdx.x * 256 + threadIdx.x;   // < B*M*128
    const int c = t & 127;
    const int m = (t >> 7) & (M_ - 1);
    const int b = t >> 17;
    const int g = c >> 4;
    const float mean = stats[b * 16 + g * 2], inv = stats[b * 16 + g * 2 + 1];
    const float ga = gamma[c] * inv, be = beta[c] - mean * ga;
    const __bf16* p = y + ((size_t)b * MK_ + (size_t)m * K_) * 128 + c;
    float mx = -3.4e38f;
#pragma unroll
    for (int k = 0; k < K_; ++k) {
        float v = (float)p[(size_t)k * 128] * ga + be;
        v = v / (1.f + __expf(-v));
        mx = fmaxf(mx, v);
    }
    out[((size_t)b * 128 + c) * M_ + m] = mx;
}

// time_emb[:, :, :M] copy
__global__ __launch_bounds__(256) void tcopy_kernel(const float* __restrict__ te,
                                                    float* __restrict__ o) {
    const int i = blockIdx.x * 256 + threadIdx.x;   // < B*64*M
    const int m = i & (M_ - 1);
    const int c = (i >> 10) & 63;
    const int b = i >> 16;
    o[i] = te[((size_t)b * 64 + c) * N_ + m];
}

// ---------------------------------------------------------------------------
extern "C" void kernel_launch(void* const* d_in, const int* in_sizes, int n_in,
                              void* d_out, int out_size, void* d_ws, size_t ws_size,
                              hipStream_t stream) {
    (void)in_sizes; (void)n_in; (void)out_size; (void)ws_size;
    const float* feat   = (const float*)d_in[0];
    const float* coords = (const float*)d_in[1];
    const float* temb   = (const float*)d_in[2];
    const float* w1 = (const float*)d_in[3],  *b1 = (const float*)d_in[4];
    const float* g1 = (const float*)d_in[5],  *be1 = (const float*)d_in[6];
    const float* w2 = (const float*)d_in[7],  *b2 = (const float*)d_in[8];
    const float* g2 = (const float*)d_in[9],  *be2 = (const float*)d_in[10];
    const float* w3 = (const float*)d_in[11], *b3 = (const float*)d_in[12];
    const float* g3 = (const float*)d_in[13], *be3 = (const float*)d_in[14];

    float* out     = (float*)d_out;                       // [16,128,1024]
    float* centers = out + (size_t)16 * 128 * 1024;       // [16,3,1024]
    float* out3    = centers + (size_t)16 * 3 * 1024;     // [16,64,1024]

    char* ws = (char*)d_ws;
    __bf16* bufA  = (__bf16*)ws;                                   // [NTOT][96]
    __bf16* bufB  = (__bf16*)(ws + (size_t)NTOT_ * 96 * 2);        // [NTOT][128]
    __bf16* featT = (__bf16*)(ws + (size_t)NTOT_ * (96 + 128) * 2);// [B*N][64]
    int*  nidx   = (int*)((char*)featT + (size_t)B_ * N_ * 64 * 2);
    float* stats = (float*)((char*)nidx + (size_t)NTOT_ * 4);      // 256 f32
    float* parts = stats + 256;                                    // 2048*16 f32

    constexpr int SMEM1 = 96 * 256 * 2 + 64 * 96 * 2 + 64 * 4 + 128;    // 61824
    constexpr int SMEM2 = 64 * 256 * 2 + 64 * 64 * 2 + 64 * 4 + 128;    // 41344
    constexpr int SMEM3 = 64 * 256 * 2 + 128 * 64 * 2 + 128 * 4 + 128;  // 49792

    featT_kernel<<<(B_ * N_ * 8) / 256, 256, 0, stream>>>(feat, featT);
    fps_kernel<<<16, 1024, 0, stream>>>(coords, centers);
    ballq_kernel<<<2048, 256, 0, stream>>>(coords, centers, nidx);
    gather_kernel<<<NTOT_ / 256, 256, 0, stream>>>(featT, coords, centers, nidx, bufA);

    // Layer 1: 67(->96) -> 64
    gemm_gn_kernel<64, 96, 67><<<2048, 256, SMEM1, stream>>>(bufA, w1, b1, bufB, parts);
    gn_finalize_kernel<<<1, 128, 0, stream>>>(parts, stats, (float)(8 * MK_));
    gn_apply_kernel<<<64 * (NTOT_ / 8) / 256, 256, 0, stream>>>(bufB, stats, g1, be1, 64, 8);

    // Layer 2: 64 -> 64
    gemm_gn_kernel<64, 64, 64><<<2048, 256, SMEM2, stream>>>(bufB, w2, b2, bufA, parts);
    gn_finalize_kernel<<<1, 128, 0, stream>>>(parts, stats, (float)(8 * MK_));
    gn_apply_kernel<<<64 * (NTOT_ / 8) / 256, 256, 0, stream>>>(bufA, stats, g2, be2, 64, 8);

    // Layer 3: 64 -> 128, then GN+SiLU+max fused
    gemm_gn_kernel<128, 64, 64><<<2048, 256, SMEM3, stream>>>(bufA, w3, b3, bufB, parts);
    gn_finalize_kernel<<<1, 128, 0, stream>>>(parts, stats, (float)(16 * MK_));
    gn_silu_max_kernel<<<(16 * 128 * 1024) / 256, 256, 0, stream>>>(bufB, stats, g3, be3, out);

    tcopy_kernel<<<(16 * 64 * 1024) / 256, 256, 0, stream>>>(temb, out3);
}
